// CeleTrip_Word_Doc_15195594293514
// MI455X (gfx1250) — matrix operand order repaired and checked by reading, not compiled, not run on hardware
//
#include <hip/hip_runtime.h>
#include <hip/hip_bf16.h>

typedef _Float16 half_t;
typedef __attribute__((ext_vector_type(16))) _Float16 v16h;
typedef __attribute__((ext_vector_type(8)))  _Float16 v8h;
typedef __attribute__((ext_vector_type(8)))  float    v8f;

#define N_DOC   5000
#define N_WORD  45000
#define NN      (N_DOC + N_WORD)     // 50000 nodes (divisible by 16)
#define NHEAD   4
#define DDIM    64
#define HD      (NHEAD * DDIM)       // 256
#define E_EDGES 850000               // 2*400000 + NN self loops
#define NEG_SLOPE 0.2f

#define N_DOC_PAD  5008              // round up to 16
#define N_WORD_PAD 45008
#define K_WORD_PAD 320               // 300 rounded up to 32

// ---------------------------------------------------------------------------
// f32 -> f16 convert with row/K zero padding:
// out[r, k] (r<rowsPad, k<Kpad) = (r<rowsReal && k<Kreal) ? in[r,k] : 0
// ---------------------------------------------------------------------------
__global__ void cvt_pad(const float* __restrict__ in, half_t* __restrict__ out,
                        long rowsReal, long rowsPad, int Kreal, int Kpad)
{
    long t = (long)blockIdx.x * blockDim.x + threadIdx.x;
    if (t >= rowsPad * Kpad) return;
    long r = t / Kpad;
    int  k = (int)(t - r * Kpad);
    float v = (r < rowsReal && k < Kreal) ? in[r * Kreal + k] : 0.f;
    out[t] = (half_t)v;
}

__global__ void fill_f32(float* __restrict__ p, long n, float v) {
    long t = (long)blockIdx.x * blockDim.x + threadIdx.x;
    if (t < n) p[t] = v;
}

// ---------------------------------------------------------------------------
// Pack W[K, Nout] (f32) into B-fragment order (f16), zero-padding K to Kpad.
// Fragment layout (16-bit B 32x16): lanes 0-15 -> K 0..15, lanes 16-31 -> K 16..31,
// element e: v=e/2, K = kc*32 + (lane>=16)*16 + 2*v + (e&1), N = ct*16 + (lane&15).
// Stored as P[(((ct*nkc)+kc)*32 + lane)*16 + e]  -> one contiguous 32B frag load.
// ---------------------------------------------------------------------------
__global__ void packB(const float* __restrict__ W, half_t* __restrict__ P,
                      int Kreal, int Kpad, int Nout)
{
    const int nkc = Kpad >> 5;
    long t = (long)blockIdx.x * blockDim.x + threadIdx.x;
    if (t >= (long)Kpad * Nout) return;
    int  e    = (int)(t & 15);
    int  lane = (int)((t >> 4) & 31);
    long tile = t >> 9;
    int  kc   = (int)(tile % nkc);
    int  ct   = (int)(tile / nkc);
    int  hi   = lane >> 4;
    int  col  = ct * 16 + (lane & 15);
    int  k    = kc * 32 + hi * 16 + (e & ~1) + (e & 1);
    float v = (k < Kreal) ? W[(long)k * Nout + col] : 0.f;
    P[t] = (half_t)v;
}

// ---------------------------------------------------------------------------
// WMMA GEMM: Y[nRows, Nout] = X[rowsPad, Kpad](f16) @ P(packed f16, K x Nout)
// X must be padded so every 16-row tile is fully readable and Kpad % 32 == 0.
// One wave per 16x16 tile; inner loop: 4x global_load_b128 + 1x v_wmma.
// ---------------------------------------------------------------------------
__global__ void gemm_wmma(const half_t* __restrict__ X, int ldx, int Kpad,
                          const half_t* __restrict__ P,
                          const float* __restrict__ bias, int doRelu,
                          float* __restrict__ Y, half_t* __restrict__ Yh,
                          int nRows, int Nout)
{
    const int lane  = threadIdx.x & 31;
    const int mrow  = lane & 15;
    const int hi    = lane >> 4;
    const int wpb   = blockDim.x >> 5;
    const int wave  = blockIdx.x * wpb + (threadIdx.x >> 5);
    const int nwave = gridDim.x * wpb;
    const int colTiles = Nout >> 4;
    const int rowTiles = (nRows + 15) >> 4;
    const int tiles = rowTiles * colTiles;
    const int nkc   = Kpad >> 5;

    for (int t = wave; t < tiles; t += nwave) {
        const int rt = t / colTiles;
        const int ct = t - rt * colTiles;
        const int arow = rt * 16 + mrow;

        const half_t* ap = X + (long)arow * ldx + hi * 16;
        const v16h*   bp = (const v16h*)(P + ((long)ct * nkc * 32 + lane) * 16);

        v8f acc = {};
        for (int kc = 0; kc < nkc; ++kc) {
            v8h alo = *(const v8h*)(ap);            // halves [k0+hi*16,  +8)
            v8h ahi = *(const v8h*)(ap + 8);        // halves [k0+hi*16+8,+8)
            v16h a;
#pragma unroll
            for (int e = 0; e < 8; ++e) { a[e] = alo[e]; a[8 + e] = ahi[e]; }
            v16h b = bp[kc * 32];                   // packed fragment, contiguous
            acc = __builtin_amdgcn_wmma_f32_16x16x32_f16(
                false, a, false, b, (short)0, acc, false, false);
            ap += 32;
        }

        const int ocol = ct * 16 + mrow;
#pragma unroll
        for (int r = 0; r < 8; ++r) {
            int orow = rt * 16 + r + hi * 8;        // D: M = r (+8 for lanes 16-31)
            if (orow < nRows) {
                float v = acc[r];
                if (bias) v += bias[ocol];
                if (doRelu) v = v > 0.f ? v : 0.f;
                Y[(long)orow * Nout + ocol] = v;
                if (Yh) Yh[(long)orow * Nout + ocol] = (half_t)v;
            }
        }
    }
}

// ---------------------------------------------------------------------------
// el[n,h] = sum_d h[n,h,d]*al[h,d] ; er likewise.  One wave per (n,h),
// 2 floats per lane, butterfly reduction (coalesced 256B loads).
// ---------------------------------------------------------------------------
__global__ void scores_kernel(const float* __restrict__ h,
                              const float* __restrict__ al, const float* __restrict__ ar,
                              float* __restrict__ el, float* __restrict__ er)
{
    int wave = (blockIdx.x * blockDim.x + threadIdx.x) >> 5;
    int lane = threadIdx.x & 31;
    if (wave >= NN * NHEAD) return;
    int n = wave >> 2, hh = wave & 3;
    const float* hp = h + (long)n * HD + hh * DDIM + lane * 2;
    const float* ap = al + hh * DDIM + lane * 2;
    const float* bp = ar + hh * DDIM + lane * 2;
    float v0 = hp[0], v1 = hp[1];
    float sl = v0 * ap[0] + v1 * ap[1];
    float sr = v0 * bp[0] + v1 * bp[1];
#pragma unroll
    for (int off = 16; off >= 1; off >>= 1) {
        sl += __shfl_xor(sl, off);
        sr += __shfl_xor(sr, off);
    }
    if (lane == 0) { el[wave] = sl; er[wave] = sr; }
}

// ---------------------------------------------------------------------------
// Edge softmax pass 1: segment max over dst
// ---------------------------------------------------------------------------
__global__ void edge_max_kernel(const int* __restrict__ src, const int* __restrict__ dst,
                                const float* __restrict__ el, const float* __restrict__ er,
                                float* __restrict__ m)
{
    int t = blockIdx.x * blockDim.x + threadIdx.x;
    if (t >= E_EDGES * NHEAD) return;
    int e = t >> 2, hh = t & 3;
    int s = src[e], d = dst[e];
    float x = el[s * NHEAD + hh] + er[d * NHEAD + hh];
    x = x > 0.f ? x : NEG_SLOPE * x;
    atomicMax(&m[d * NHEAD + hh], x);              // global_atomic_max_num_f32
}

// Pass 2: ex = exp(e - m[dst]); segment sum
__global__ void edge_exp_kernel(const int* __restrict__ src, const int* __restrict__ dst,
                                const float* __restrict__ el, const float* __restrict__ er,
                                const float* __restrict__ m,
                                float* __restrict__ exbuf, float* __restrict__ ssum)
{
    int t = blockIdx.x * blockDim.x + threadIdx.x;
    if (t >= E_EDGES * NHEAD) return;
    int e = t >> 2, hh = t & 3;
    int s = src[e], d = dst[e];
    float x = el[s * NHEAD + hh] + er[d * NHEAD + hh];
    x = x > 0.f ? x : NEG_SLOPE * x;
    float ex = expf(x - m[d * NHEAD + hh]);
    exbuf[t] = ex;
    atomicAdd(&ssum[d * NHEAD + hh], ex);
}

// Pass 3: acc[dst, c] += (ex/s[dst]) * h[src, c]   -- one wave per edge
__global__ void edge_agg_kernel(const int* __restrict__ src, const int* __restrict__ dst,
                                const float* __restrict__ exbuf, const float* __restrict__ ssum,
                                const float* __restrict__ h, float* __restrict__ acc)
{
    int wave = (blockIdx.x * blockDim.x + threadIdx.x) >> 5;
    int lane = threadIdx.x & 31;
    if (wave >= E_EDGES) return;
    int s = src[wave], d = dst[wave];
    float a[NHEAD];
#pragma unroll
    for (int hh = 0; hh < NHEAD; ++hh)
        a[hh] = exbuf[wave * NHEAD + hh] / ssum[d * NHEAD + hh];
#pragma unroll
    for (int i = 0; i < 8; ++i) {
        int c = lane + i * 32;                     // coalesced across the wave
        float v = a[c >> 6] * h[(long)s * HD + c];
        atomicAdd(&acc[(long)d * HD + c], v);
    }
}

// x = relu(acc + res + bias); write f32 (+ optional f16 for next GEMM)
__global__ void finalize_kernel(const float* __restrict__ acc, const float* __restrict__ res,
                                const float* __restrict__ bias,
                                float* __restrict__ xout, half_t* __restrict__ xh)
{
    long t = (long)blockIdx.x * blockDim.x + threadIdx.x;
    if (t >= (long)NN * HD) return;
    int c = (int)(t & (HD - 1));
    float v = acc[t] + res[t] + bias[c];
    v = v > 0.f ? v : 0.f;
    xout[t] = v;
    if (xh) xh[t] = (half_t)v;
}

// Max-pool over nodes: pooled[c] = max_n x[n,c]
__global__ void pool_kernel(const float* __restrict__ x, float* __restrict__ pooled)
{
    int c = threadIdx.x;                           // 256 threads
    float mx = -1e30f;
    for (int n = blockIdx.x; n < NN; n += gridDim.x)
        mx = fmaxf(mx, x[(long)n * HD + c]);
    atomicMax(&pooled[c], mx);
}

// head: out = relu(pooled @ Ws + bs) @ Wc + bc   (single block, tiny)
__global__ void head_kernel(const float* __restrict__ pooled,
                            const float* __restrict__ Ws, const float* __restrict__ bs,
                            const float* __restrict__ Wc, const float* __restrict__ bc,
                            float* __restrict__ out)
{
    __shared__ float tbuf[128];
    int j = threadIdx.x;                           // 128 threads
    float s = bs[j];
    for (int c = 0; c < HD; ++c) s += pooled[c] * Ws[c * 128 + j];
    tbuf[j] = s > 0.f ? s : 0.f;
    __syncthreads();
    if (j < 2) {
        float o = bc[j];
        for (int k = 0; k < 128; ++k) o += tbuf[k] * Wc[k * 2 + j];
        out[j] = o;
    }
}

// ---------------------------------------------------------------------------
static inline int cdiv(long n, int b) { return (int)((n + b - 1) / b); }

extern "C" void kernel_launch(void* const* d_in, const int* in_sizes, int n_in,
                              void* d_out, int out_size, void* d_ws, size_t ws_size,
                              hipStream_t stream)
{
    const float* doc_feat  = (const float*)d_in[0];   // [5000,768]
    const float* word_feat = (const float*)d_in[1];   // [45000,300]
    const int*   src       = (const int*)d_in[2];     // [E]
    const int*   dst       = (const int*)d_in[3];     // [E]
    const float* Wd        = (const float*)d_in[4];   // [768,64]
    const float* bd        = (const float*)d_in[5];   // [64]
    const float* Ww        = (const float*)d_in[6];   // [300,64]
    const float* bw        = (const float*)d_in[7];   // [64]
    const float* W1        = (const float*)d_in[8];   // [64,256]
    const float* al1       = (const float*)d_in[9];   // [4,64]
    const float* ar1       = (const float*)d_in[10];  // [4,64]
    const float* b1        = (const float*)d_in[11];  // [256]
    const float* resW1     = (const float*)d_in[12];  // [64,256]
    const float* W2        = (const float*)d_in[13];  // [256,256]
    const float* al2       = (const float*)d_in[14];
    const float* ar2       = (const float*)d_in[15];
    const float* b2        = (const float*)d_in[16];  // [256]
    const float* Ws        = (const float*)d_in[17];  // [256,128]
    const float* bs        = (const float*)d_in[18];  // [128]
    const float* Wc        = (const float*)d_in[19];  // [128,2]
    const float* bc        = (const float*)d_in[20];  // [2]

    // ---- workspace bump allocator -------------------------------------------------
    char* wp = (char*)d_ws;
    auto alloc = [&](size_t bytes) -> void* {
        void* p = (void*)wp;
        wp += (bytes + 255) & ~(size_t)255;
        return p;
    };
    half_t* doc_h   = (half_t*)alloc((size_t)N_DOC_PAD  * 768 * sizeof(half_t));
    half_t* word_h  = (half_t*)alloc((size_t)N_WORD_PAD * K_WORD_PAD * sizeof(half_t));
    half_t* PWd     = (half_t*)alloc((size_t)768 * 64  * sizeof(half_t));
    half_t* PWw     = (half_t*)alloc((size_t)K_WORD_PAD * 64 * sizeof(half_t));
    half_t* PW1     = (half_t*)alloc((size_t)64  * 256 * sizeof(half_t));
    half_t* PrW1    = (half_t*)alloc((size_t)64  * 256 * sizeof(half_t));
    half_t* PW2     = (half_t*)alloc((size_t)256 * 256 * sizeof(half_t));
    float*  x0      = (float*) alloc((size_t)NN * 64  * sizeof(float));
    half_t* xh      = (half_t*)alloc((size_t)NN * HD  * sizeof(half_t)); // [NN,64] then [NN,256]
    float*  hbuf    = (float*) alloc((size_t)NN * HD  * sizeof(float));
    float*  resbuf  = (float*) alloc((size_t)NN * HD  * sizeof(float));
    float*  accbuf  = (float*) alloc((size_t)NN * HD  * sizeof(float));
    float*  x1      = (float*) alloc((size_t)NN * HD  * sizeof(float));
    float*  x2      = (float*) alloc((size_t)NN * HD  * sizeof(float));
    float*  el      = (float*) alloc((size_t)NN * NHEAD * sizeof(float));
    float*  er      = (float*) alloc((size_t)NN * NHEAD * sizeof(float));
    float*  mbuf    = (float*) alloc((size_t)NN * NHEAD * sizeof(float));
    float*  sbuf    = (float*) alloc((size_t)NN * NHEAD * sizeof(float));
    float*  exbuf   = (float*) alloc((size_t)E_EDGES * NHEAD * sizeof(float));
    float*  pooled  = (float*) alloc((size_t)HD * sizeof(float));
    (void)ws_size; (void)n_in; (void)in_sizes; (void)out_size;

    const int B = 256;

    // ---- f16 conversions (with padding) + weight packing --------------------------
    cvt_pad<<<cdiv((long)N_DOC_PAD * 768, B), B, 0, stream>>>(
        doc_feat, doc_h, N_DOC, N_DOC_PAD, 768, 768);
    cvt_pad<<<cdiv((long)N_WORD_PAD * K_WORD_PAD, B), B, 0, stream>>>(
        word_feat, word_h, N_WORD, N_WORD_PAD, 300, K_WORD_PAD);
    packB<<<cdiv(768 * 64, B), B, 0, stream>>>(Wd,    PWd,  768, 768, 64);
    packB<<<cdiv(K_WORD_PAD * 64, B), B, 0, stream>>>(Ww,   PWw,  300, K_WORD_PAD, 64);
    packB<<<cdiv(64 * 256, B), B, 0, stream>>>(W1,    PW1,  64,  64,  256);
    packB<<<cdiv(64 * 256, B), B, 0, stream>>>(resW1, PrW1, 64,  64,  256);
    packB<<<cdiv(256 * 256, B), B, 0, stream>>>(W2,   PW2,  256, 256, 256);

    // ---- input projections: x0 = relu([doc;word] @ {Wd,Ww} + {bd,bw}) --------------
    gemm_wmma<<<512, 256, 0, stream>>>(doc_h, 768, 768, PWd, bd, 1,
                                       x0, xh, N_DOC, 64);
    gemm_wmma<<<512, 256, 0, stream>>>(word_h, K_WORD_PAD, K_WORD_PAD, PWw, bw, 1,
                                       x0 + (long)N_DOC * 64, xh + (long)N_DOC * 64,
                                       N_WORD, 64);

    const long nodeElems = (long)NN * HD;
    const int  eh        = E_EDGES * NHEAD;

    // =========================== GAT layer 1 =======================================
    gemm_wmma<<<512, 256, 0, stream>>>(xh, 64, 64, PW1,  nullptr, 0, hbuf,   nullptr, NN, 256);
    gemm_wmma<<<512, 256, 0, stream>>>(xh, 64, 64, PrW1, nullptr, 0, resbuf, nullptr, NN, 256);
    scores_kernel<<<cdiv((long)NN * NHEAD * 32, B), B, 0, stream>>>(hbuf, al1, ar1, el, er);
    fill_f32<<<cdiv(NN * NHEAD, B), B, 0, stream>>>(mbuf, NN * NHEAD, -1e30f);
    fill_f32<<<cdiv(NN * NHEAD, B), B, 0, stream>>>(sbuf, NN * NHEAD, 0.f);
    fill_f32<<<cdiv(nodeElems, B), B, 0, stream>>>(accbuf, nodeElems, 0.f);
    edge_max_kernel<<<cdiv(eh, B), B, 0, stream>>>(src, dst, el, er, mbuf);
    edge_exp_kernel<<<cdiv(eh, B), B, 0, stream>>>(src, dst, el, er, mbuf, exbuf, sbuf);
    edge_agg_kernel<<<cdiv((long)E_EDGES * 32, B), B, 0, stream>>>(src, dst, exbuf, sbuf,
                                                                   hbuf, accbuf);
    finalize_kernel<<<cdiv(nodeElems, B), B, 0, stream>>>(accbuf, resbuf, b1, x1, xh);

    // =========================== GAT layer 2 (identity residual) ===================
    gemm_wmma<<<512, 256, 0, stream>>>(xh, 256, 256, PW2, nullptr, 0, hbuf, nullptr, NN, 256);
    scores_kernel<<<cdiv((long)NN * NHEAD * 32, B), B, 0, stream>>>(hbuf, al2, ar2, el, er);
    fill_f32<<<cdiv(NN * NHEAD, B), B, 0, stream>>>(mbuf, NN * NHEAD, -1e30f);
    fill_f32<<<cdiv(NN * NHEAD, B), B, 0, stream>>>(sbuf, NN * NHEAD, 0.f);
    fill_f32<<<cdiv(nodeElems, B), B, 0, stream>>>(accbuf, nodeElems, 0.f);
    edge_max_kernel<<<cdiv(eh, B), B, 0, stream>>>(src, dst, el, er, mbuf);
    edge_exp_kernel<<<cdiv(eh, B), B, 0, stream>>>(src, dst, el, er, mbuf, exbuf, sbuf);
    edge_agg_kernel<<<cdiv((long)E_EDGES * 32, B), B, 0, stream>>>(src, dst, exbuf, sbuf,
                                                                   hbuf, accbuf);
    finalize_kernel<<<cdiv(nodeElems, B), B, 0, stream>>>(accbuf, x1, b2, x2, nullptr);

    // ---- max-pool over nodes + MLP head -------------------------------------------
    fill_f32<<<1, HD, 0, stream>>>(pooled, HD, -1e30f);
    pool_kernel<<<256, HD, 0, stream>>>(x2, pooled);
    head_kernel<<<1, 128, 0, stream>>>(pooled, Ws, bs, Wc, bc, (float*)d_out);
}
